// RSSM_240518168920
// MI455X (gfx1250) — matrix operand order repaired and checked by reading, not compiled, not run on hardware
//
#include <hip/hip_runtime.h>

// ---------------------------------------------------------------------------
// RSSM scan for MI455X (gfx1250, wave32, WMMA).
// Parallel precompute: embed @ po_w0_embed^T  (big GEMM, many blocks, WMMA bf16)
// Sequential chain:    1 persistent workgroup (16 wave32s), state in LDS,
//                      all matmuls via v_wmma_f32_16x16x32_bf16,
//                      one-hot stoch GEMM synthesized from category indices.
// ---------------------------------------------------------------------------

typedef __attribute__((ext_vector_type(16))) __bf16 v16bf;
typedef __attribute__((ext_vector_type(8)))  float  v8f;
typedef __attribute__((ext_vector_type(16))) unsigned short v16us;

union AFrag { v16bf v; __bf16 e[16]; };
union CFrag { v8f  v; float  e[8];  };

#define NWAVES 16           // 512 threads / wave32
#define BB 64               // batch
#define TT 512              // time

// ---- ws layout (bytes, all 256-aligned) -----------------------------------
#define OFF_WHH   0u                       // [608][224] bf16
#define OFF_PR0   272384u                  // [208][224] bf16
#define OFF_PR1   365568u
#define OFF_PR2   458752u                  // [960][224] bf16
#define OFF_PO0D  888832u                  // [208][224] bf16
#define OFF_PO1   982016u
#define OFF_PO2   1075200u                 // [960][224] bf16
#define OFF_PO0E  1505280u                 // [208][1024] bf16
#define OFF_WIH   1931264u                 // [608][960] bf16
#define OFF_WIHA  3098624u                 // [6][608] f32
#define OFF_EPRE  3113472u                 // [B*T][208] f32

// ---------------------------------------------------------------------------
// Fragment loaders (wave32; A and B share contiguous k = e + 16h order)
// A 16x32 bf16: lane m=l&15; VGPR j: K = 2*j + (l&16)
// B 32x16 bf16: lane n=l&15; elements e: K = e + (l&16)  (16 contiguous)
// C 16x16 f32:  VGPR v: M = v + ((l>>4)<<3); N = l&15
// ---------------------------------------------------------------------------
__device__ __forceinline__ v16bf afrag_ldsbf(const __bf16* A, int stride,
                                             int mBase, int kB, int lane) {
  AFrag f;
  int m  = mBase + (lane & 15);
  int lh = (lane & 16) >> 1;
#pragma unroll
  for (int j = 0; j < 8; ++j) {
    int k0 = kB + 2 * j + 2 * lh;
    f.e[2 * j]     = A[m * stride + k0];
    f.e[2 * j + 1] = A[m * stride + k0 + 1];
  }
  return f.v;
}

__device__ __forceinline__ v16bf afrag_onehot(const int* idxL,
                                              int mBase, int kB, int lane) {
  AFrag f;
  int m  = mBase + (lane & 15);
  int lh = (lane & 16) >> 1;
#pragma unroll
  for (int j = 0; j < 8; ++j) {
    int k0 = kB + 2 * j + 2 * lh;
    int s  = k0 >> 5;                    // category group (30 groups of 32)
    int ii = idxL[m * 32 + s];           // -1 => zeroed stoch (is_first)
    f.e[2 * j]     = (__bf16)((ii == (k0 & 31))       ? 1.0f : 0.0f);
    f.e[2 * j + 1] = (__bf16)((ii == ((k0 + 1) & 31)) ? 1.0f : 0.0f);
  }
  return f.v;
}

__device__ __forceinline__ v16bf bfrag_global(const unsigned short* W, int Kpad,
                                              int n, int kB, int lane) {
  // rows are 32B-aligned (Kpad multiple of 16 bf16); one aligned 32B load
  const v16us* p = (const v16us*)(W + (size_t)n * Kpad + kB + (lane & 16));
  v16us u = *p;
  return __builtin_bit_cast(v16bf, u);
}

// ---------------------------------------------------------------------------
// Generic per-step GEMM:  C[64 x Npad] = A[64 x Kpad] * W^T  (+init), epilogue
// ---------------------------------------------------------------------------
enum { AM_LDS = 0, AM_ONEHOT = 1 };
enum { IN_BIAS = 0, IN_LDSBF = 1, IN_EPRE = 2 };
enum { OUT_LDSBF = 0, OUT_GLOB = 1 };
enum { ACT_NONE = 0, ACT_ELU = 1 };

__device__ void gemm_step(int wave, int lane,
                          int aMode, const __bf16* A, int aStride, const int* idxL,
                          const unsigned short* W, int Kpad, int kSteps, int colTiles,
                          int initMode, const float* bias, const __bf16* initBf,
                          int initStride, const float* epre, int t,
                          int outMode, __bf16* outBf, int outStride, float* outG,
                          int nValid, int act) {
  int nTiles = 4 * colTiles;                 // 4 row tiles (64 rows)
  for (int tile = wave; tile < nTiles; tile += NWAVES) {
    int rt = tile & 3, ct = tile >> 2;
    int mBase = rt << 4, nBase = ct << 4;
    int n    = nBase + (lane & 15);
    int mOff = (lane >> 4) << 3;
    CFrag c;
#pragma unroll
    for (int v = 0; v < 8; ++v) {
      int m = mBase + v + mOff;
      float iv = 0.f;
      if (initMode == IN_BIAS)       { if (n < nValid) iv = bias[n]; }
      else if (initMode == IN_LDSBF) { if (n < nValid) iv = (float)initBf[m * initStride + n]; }
      else                           { iv = epre[((size_t)m * TT + t) * 208 + n]; }
      c.e[v] = iv;
    }
    for (int kk = 0; kk < kSteps; ++kk) {
      if (kk + 1 < kSteps)
        __builtin_prefetch(W + (size_t)n * Kpad + (kk + 1) * 32, 0, 1);
      v16bf a = (aMode == AM_ONEHOT)
                    ? afrag_onehot(idxL, mBase, kk * 32, lane)
                    : afrag_ldsbf(A, aStride, mBase, kk * 32, lane);
      v16bf b = bfrag_global(W, Kpad, n, kk * 32, lane);
      c.v = __builtin_amdgcn_wmma_f32_16x16x32_bf16(false, a, false, b,
                                                    (short)0, c.v, false, false);
    }
#pragma unroll
    for (int v = 0; v < 8; ++v) {
      int m = mBase + v + mOff;
      float x = c.e[v];
      if (act == ACT_ELU) x = (x > 0.f) ? x : expm1f(x);
      if (n < nValid) {
        if (outMode == OUT_LDSBF) outBf[m * outStride + n] = (__bf16)x;
        else                      outG[((size_t)m * TT + t) * 960 + n] = x;
      }
    }
  }
}

// ---------------------------------------------------------------------------
// softmax -> unimix -> log -> (write logits), gumbel-argmax -> one-hot
// ---------------------------------------------------------------------------
__device__ void softmax_pass(float* logits, const float* gum, float* stoch,
                             int* idxL, int t, int tid) {
  for (int g = tid; g < BB * 30; g += 512) {
    int b = g / 30, s = g % 30;
    size_t base = ((size_t)b * TT + t) * 960 + (size_t)s * 32;
    float v[32];
    float mx = -1e30f;
#pragma unroll
    for (int k = 0; k < 32; ++k) { v[k] = logits[base + k]; mx = fmaxf(mx, v[k]); }
    float sum = 0.f;
#pragma unroll
    for (int k = 0; k < 32; ++k) { v[k] = expf(v[k] - mx); sum += v[k]; }
    float inv = 1.f / sum;
    int arg = 0; float best = -1e30f;
#pragma unroll
    for (int k = 0; k < 32; ++k) {
      float p  = 0.99f * v[k] * inv + 0.01f / 32.f;
      float lp = logf(p);
      logits[base + k] = lp;
      stoch[base + k]  = 0.f;
      float sc = lp + gum[base + k];
      if (sc > best) { best = sc; arg = k; }
    }
    stoch[base + arg] = 1.f;
    if (idxL) idxL[b * 32 + s] = arg;
  }
}

// ---------------------------------------------------------------------------
// Prep kernels: f32 weights -> zero-padded bf16 [Npad][Kpad] in ws
// ---------------------------------------------------------------------------
__global__ void cvt_pad_bf16(const float* src, int N, int Ksrc, int c0, int Kc,
                             unsigned short* dst, int Npad, int Kpad) {
  int i = blockIdx.x * 256 + threadIdx.x;
  if (i >= Npad * Kpad) return;
  int n = i / Kpad, k = i % Kpad;
  float v = (n < N && k < Kc) ? src[(size_t)n * Ksrc + c0 + k] : 0.f;
  __bf16 b = (__bf16)v;
  dst[i] = __builtin_bit_cast(unsigned short, b);
}

__global__ void wihA_kernel(const float* w_ih, float* wihA) {
  int i = blockIdx.x * 256 + threadIdx.x;       // [6][608]
  if (i >= 6 * 608) return;
  int j = i / 608, n = i % 608;
  wihA[i] = (n < 600) ? w_ih[(size_t)n * 966 + 960 + j] : 0.f;
}

// ---------------------------------------------------------------------------
// Parallel precompute: epre[r][208] = embed[r] @ po_w0[:,200:1224]^T + po_b0
// One 16-row tile per block, embed tile staged to LDS as bf16.
// ---------------------------------------------------------------------------
__global__ __launch_bounds__(256) void epre_kernel(const float* embed,
                                                   const unsigned short* po0e,
                                                   const float* po_b0,
                                                   float* epre) {
  __shared__ __bf16 et[16 * 1024];
  int rBase = blockIdx.x * 16;
  int tid = threadIdx.x;
  for (int i = tid; i < 16 * 1024; i += 256)
    et[i] = (__bf16)embed[(size_t)rBase * 1024 + i];
  __syncthreads();
  int wave = tid >> 5, lane = tid & 31;
  for (int ct = wave; ct < 13; ct += 8) {        // 208 cols
    int n    = ct * 16 + (lane & 15);
    int mOff = (lane >> 4) << 3;
    CFrag c;
#pragma unroll
    for (int v = 0; v < 8; ++v) c.e[v] = (n < 200) ? po_b0[n] : 0.f;
    for (int kk = 0; kk < 1024; kk += 32) {
      v16bf a = afrag_ldsbf(et, 1024, 0, kk, lane);
      v16bf b = bfrag_global(po0e, 1024, n, kk, lane);
      c.v = __builtin_amdgcn_wmma_f32_16x16x32_bf16(false, a, false, b,
                                                    (short)0, c.v, false, false);
    }
#pragma unroll
    for (int v = 0; v < 8; ++v) {
      int m = v + mOff;
      epre[(size_t)(rBase + m) * 208 + n] = c.e[v];
    }
  }
}

// ---------------------------------------------------------------------------
// Persistent sequential kernel: 1 workgroup, 512 threads, T=512 steps.
// LDS: deter/actA/actB bf16 [64][224], gi/gh bf16 [64][608], idx, isf, a.
// ---------------------------------------------------------------------------
#define SMEM_SEQ (3 * 64 * 224 * 2 + 2 * 64 * 608 * 2 + 64 * 32 * 4 + 64 * 4 + 64 * 6 * 4)

__global__ __launch_bounds__(512, 1) void rssm_seq(
    const float* action, const int* is_first,
    const float* g_prior, const float* g_post,
    const float* b_ih, const float* b_hh,
    const float* pr_b0, const float* pr_b1, const float* pr_b2,
    const float* po_b1, const float* po_b2,
    const unsigned short* Wwhh, const unsigned short* Wwih,
    const unsigned short* Wpr0, const unsigned short* Wpr1, const unsigned short* Wpr2,
    const unsigned short* Wpo0d, const unsigned short* Wpo1, const unsigned short* Wpo2,
    const float* wihA, const float* epre,
    float* out_pos_stoch, float* out_deter1, float* out_pos_logits,
    float* out_pri_stoch, float* out_deter2, float* out_pri_logits) {
  extern __shared__ char smem[];
  __bf16* deterL = (__bf16*)smem;            // [64][224]
  __bf16* actA   = deterL + 64 * 224;        // [64][224]
  __bf16* actB   = actA + 64 * 224;          // [64][224]
  __bf16* giL    = actB + 64 * 224;          // [64][608]
  __bf16* ghL    = giL + 64 * 608;           // [64][608]
  int*    idxL   = (int*)(ghL + 64 * 608);   // [64][32]
  int*    isfL   = idxL + 64 * 32;           // [64]
  float*  aL     = (float*)(isfL + 64);      // [64][6]

  int tid = threadIdx.x;
  int wave = tid >> 5, lane = tid & 31;

  // init state: deter=0, stoch one-hot at index 0, activation padding zero
  for (int i = tid; i < 64 * 224; i += 512) {
    deterL[i] = (__bf16)0.f; actA[i] = (__bf16)0.f; actB[i] = (__bf16)0.f;
  }
  for (int i = tid; i < 64 * 32; i += 512) idxL[i] = 0;
  __syncthreads();

  for (int t = 0; t < TT; ++t) {
    // ---- stage per-step scalars ----
    if (tid < 64)  isfL[tid] = is_first[tid * TT + t];
    if (tid < 384) aL[tid] = action[(size_t)((tid / 6) * TT + t) * 6 + (tid % 6)];
    __syncthreads();

    // ---- is_first reset + action-bias pre-pass (independent writes) ----
    for (int i = tid; i < 64 * 224; i += 512)
      if (isfL[i / 224]) deterL[i] = (__bf16)0.f;
    for (int i = tid; i < 64 * 32; i += 512)
      if (isfL[i / 32]) idxL[i] = -1;                 // zeroed stoch
    for (int i = tid; i < 64 * 600; i += 512) {       // gi := b_ih + a @ w_ihA^T
      int b = i / 600, n = i % 600;
      float v = b_ih[n];
#pragma unroll
      for (int j = 0; j < 6; ++j) v += aL[b * 6 + j] * wihA[j * 608 + n];
      giL[b * 608 + n] = (__bf16)v;
    }
    __syncthreads();

    // ---- GRU input GEMMs (WMMA) ----
    // gi += onehot(stoch) @ w_ih[:, :960]^T   (A synthesized from indices)
    gemm_step(wave, lane, AM_ONEHOT, nullptr, 0, idxL, Wwih, 960, 30, 38,
              IN_LDSBF, nullptr, giL, 608, nullptr, t,
              OUT_LDSBF, giL, 608, nullptr, 600, ACT_NONE);
    // gh = deter @ w_hh^T + b_hh
    gemm_step(wave, lane, AM_LDS, deterL, 224, nullptr, Wwhh, 224, 7, 38,
              IN_BIAS, b_hh, nullptr, 0, nullptr, t,
              OUT_LDSBF, ghL, 608, nullptr, 600, ACT_NONE);
    __syncthreads();

    // ---- GRU gates ----
    for (int i = tid; i < 64 * 200; i += 512) {
      int b = i / 200, d = i % 200;
      float gir = (float)giL[b * 608 + d];
      float giz = (float)giL[b * 608 + 200 + d];
      float gin = (float)giL[b * 608 + 400 + d];
      float ghr = (float)ghL[b * 608 + d];
      float ghz = (float)ghL[b * 608 + 200 + d];
      float ghn = (float)ghL[b * 608 + 400 + d];
      float old = (float)deterL[b * 224 + d];
      float r  = 1.f / (1.f + expf(-(gir + ghr)));
      float z  = 1.f / (1.f + expf(-(giz + ghz)));
      float nn = tanhf(gin + r * ghn);
      float nd = (1.f - z) * nn + z * old;
      deterL[b * 224 + d] = (__bf16)nd;
      size_t o = (size_t)(b * TT + t) * 200 + d;
      out_deter1[o] = nd;
      out_deter2[o] = nd;
    }
    __syncthreads();

    // ---- prior head ----
    gemm_step(wave, lane, AM_LDS, deterL, 224, nullptr, Wpr0, 224, 7, 13,
              IN_BIAS, pr_b0, nullptr, 0, nullptr, t,
              OUT_LDSBF, actA, 224, nullptr, 200, ACT_ELU);
    __syncthreads();
    gemm_step(wave, lane, AM_LDS, actA, 224, nullptr, Wpr1, 224, 7, 13,
              IN_BIAS, pr_b1, nullptr, 0, nullptr, t,
              OUT_LDSBF, actB, 224, nullptr, 200, ACT_ELU);
    __syncthreads();
    gemm_step(wave, lane, AM_LDS, actB, 224, nullptr, Wpr2, 224, 7, 60,
              IN_BIAS, pr_b2, nullptr, 0, nullptr, t,
              OUT_GLOB, nullptr, 0, out_pri_logits, 960, ACT_NONE);
    __threadfence();
    __syncthreads();
    softmax_pass(out_pri_logits, g_prior, out_pri_stoch, nullptr, t, tid);
    __syncthreads();

    // ---- posterior head (embed part precomputed into epre) ----
    gemm_step(wave, lane, AM_LDS, deterL, 224, nullptr, Wpo0d, 224, 7, 13,
              IN_EPRE, nullptr, nullptr, 0, epre, t,
              OUT_LDSBF, actA, 224, nullptr, 200, ACT_ELU);
    __syncthreads();
    gemm_step(wave, lane, AM_LDS, actA, 224, nullptr, Wpo1, 224, 7, 13,
              IN_BIAS, po_b1, nullptr, 0, nullptr, t,
              OUT_LDSBF, actB, 224, nullptr, 200, ACT_ELU);
    __syncthreads();
    gemm_step(wave, lane, AM_LDS, actB, 224, nullptr, Wpo2, 224, 7, 60,
              IN_BIAS, po_b2, nullptr, 0, nullptr, t,
              OUT_GLOB, nullptr, 0, out_pos_logits, 960, ACT_NONE);
    __threadfence();
    __syncthreads();
    softmax_pass(out_pos_logits, g_post, out_pos_stoch, idxL, t, tid);
    __syncthreads();   // idxL visible for next step's one-hot GEMM
  }
}

// ---------------------------------------------------------------------------
extern "C" void kernel_launch(void* const* d_in, const int* in_sizes, int n_in,
                              void* d_out, int out_size, void* d_ws, size_t ws_size,
                              hipStream_t stream) {
  (void)in_sizes; (void)n_in; (void)out_size; (void)ws_size;
  const float* embed    = (const float*)d_in[0];
  const float* action   = (const float*)d_in[1];
  const int*   is_first = (const int*)d_in[2];
  const float* g_prior  = (const float*)d_in[3];
  const float* g_post   = (const float*)d_in[4];
  const float* w_ih  = (const float*)d_in[5];
  const float* w_hh  = (const float*)d_in[6];
  const float* b_ih  = (const float*)d_in[7];
  const float* b_hh  = (const float*)d_in[8];
  const float* pr_w0 = (const float*)d_in[9];
  const float* pr_b0 = (const float*)d_in[10];
  const float* pr_w1 = (const float*)d_in[11];
  const float* pr_b1 = (const float*)d_in[12];
  const float* pr_w2 = (const float*)d_in[13];
  const float* pr_b2 = (const float*)d_in[14];
  const float* po_w0 = (const float*)d_in[15];
  const float* po_b0 = (const float*)d_in[16];
  const float* po_w1 = (const float*)d_in[17];
  const float* po_b1 = (const float*)d_in[18];
  const float* po_w2 = (const float*)d_in[19];
  const float* po_b2 = (const float*)d_in[20];

  char* ws = (char*)d_ws;
  unsigned short* WHH  = (unsigned short*)(ws + OFF_WHH);
  unsigned short* PR0  = (unsigned short*)(ws + OFF_PR0);
  unsigned short* PR1  = (unsigned short*)(ws + OFF_PR1);
  unsigned short* PR2  = (unsigned short*)(ws + OFF_PR2);
  unsigned short* PO0D = (unsigned short*)(ws + OFF_PO0D);
  unsigned short* PO1  = (unsigned short*)(ws + OFF_PO1);
  unsigned short* PO2  = (unsigned short*)(ws + OFF_PO2);
  unsigned short* PO0E = (unsigned short*)(ws + OFF_PO0E);
  unsigned short* WIH  = (unsigned short*)(ws + OFF_WIH);
  float* WIHA = (float*)(ws + OFF_WIHA);
  float* EPRE = (float*)(ws + OFF_EPRE);

  auto cvt = [&](const float* src, int N, int Ks, int c0, int Kc,
                 unsigned short* dst, int Npad, int Kpad) {
    int tot = Npad * Kpad;
    cvt_pad_bf16<<<(tot + 255) / 256, 256, 0, stream>>>(src, N, Ks, c0, Kc,
                                                        dst, Npad, Kpad);
  };
  cvt(w_hh, 600, 200, 0, 200, WHH, 608, 224);
  cvt(pr_w0, 200, 200, 0, 200, PR0, 208, 224);
  cvt(pr_w1, 200, 200, 0, 200, PR1, 208, 224);
  cvt(pr_w2, 960, 200, 0, 200, PR2, 960, 224);
  cvt(po_w0, 200, 1224, 0, 200, PO0D, 208, 224);
  cvt(po_w1, 200, 200, 0, 200, PO1, 208, 224);
  cvt(po_w2, 960, 200, 0, 200, PO2, 960, 224);
  cvt(po_w0, 200, 1224, 200, 1024, PO0E, 208, 1024);
  cvt(w_ih, 600, 966, 0, 960, WIH, 608, 960);
  wihA_kernel<<<(6 * 608 + 255) / 256, 256, 0, stream>>>(w_ih, WIHA);

  // parallel: posterior layer0 embed contribution for all (b,t)
  epre_kernel<<<2048, 256, 0, stream>>>(embed, PO0E, po_b0, EPRE);

  float* out = (float*)d_out;
  const size_t SZL = 64ull * 512 * 960;   // logits / stoch block
  const size_t SZD = 64ull * 512 * 200;   // deter block
  rssm_seq<<<1, 512, SMEM_SEQ, stream>>>(
      action, is_first, g_prior, g_post,
      b_ih, b_hh, pr_b0, pr_b1, pr_b2, po_b1, po_b2,
      WHH, WIH, PR0, PR1, PR2, PO0D, PO1, PO2, WIHA, EPRE,
      out,                       // pos_stoch
      out + SZL,                 // deter (post)
      out + SZL + SZD,           // pos_logits
      out + 2 * SZL + SZD,       // pri_stoch
      out + 3 * SZL + SZD,       // deter (prior)
      out + 3 * SZL + 2 * SZD);  // pri_logits
}
